// CrossAttentionBlock_39840116637866
// MI455X (gfx1250) — matrix operand order repaired and checked by reading, not compiled, not run on hardware
//
#include <hip/hip_runtime.h>
#include <hip/hip_bf16.h>

// CrossAttentionBlock for MI455X (gfx1250): f16 WMMA GEMMs + flash attention.
// Requires ws_size >= ~57.2 MB (6 activation buffers of 9.44MB + 4 f16 weights).

#define C_DIM 256
#define NTOK  2304        // 48*48
#define BATCH 8
#define ATT_SCALE 0.17677669529663687f   // (256/8)^-0.5
#define LOG2E 1.44269504088896f

typedef _Float16 h16 __attribute__((ext_vector_type(16)));
typedef _Float16 h8  __attribute__((ext_vector_type(8)));
typedef _Float16 h4  __attribute__((ext_vector_type(4)));
typedef float    f8  __attribute__((ext_vector_type(8)));
typedef float    f4  __attribute__((ext_vector_type(4)));

__device__ __forceinline__ f8 wmma16(h16 a, h16 b, f8 c) {
  // v_wmma_f32_16x16x32_f16 : D = A(16x32) * B(32x16) + C
  return __builtin_amdgcn_wmma_f32_16x16x32_f16(false, a, false, b, (short)0, c,
                                                false, false);
}

// A-fragment (16x32 f16, M = lane%16): K halves aoff..aoff+15 contiguous,
// aoff = lane<16 ? 0 : 16.  p = row base + k0, must be 16B aligned.
__device__ __forceinline__ h16 load_a_frag(const _Float16* p) {
  const int aoff = ((threadIdx.x & 31) < 16) ? 0 : 16;
  h8 lo = *(const h8*)(p + aoff);
  h8 hi = *(const h8*)(p + 8 + aoff);
  h16 r;
#pragma unroll
  for (int i = 0; i < 8; ++i) { r[i] = lo[i]; r[i + 8] = hi[i]; }
  return r;
}

// Same A pattern but only 8B-aligned source (padded LDS rows).
__device__ __forceinline__ h16 load_a_frag_u8(const _Float16* p) {
  const int aoff = ((threadIdx.x & 31) < 16) ? 0 : 16;
  h4 a0 = *(const h4*)(p + aoff);
  h4 a1 = *(const h4*)(p + aoff + 4);
  h4 a2 = *(const h4*)(p + 8 + aoff);
  h4 a3 = *(const h4*)(p + 8 + aoff + 4);
  h16 r;
#pragma unroll
  for (int i = 0; i < 4; ++i) {
    r[i] = a0[i]; r[4 + i] = a1[i]; r[8 + i] = a2[i]; r[12 + i] = a3[i];
  }
  return r;
}

// B-fragment (32x16 f16, N = lane%16): K halves klo..klo+15 contiguous,
// klo = lane<16 ? 0 : 16.  p = "column-row" base + k0 (row-major over K).
__device__ __forceinline__ h16 load_b_frag(const _Float16* p) {
  const int klo = ((threadIdx.x & 31) < 16) ? 0 : 16;
  return *(const h16*)(p + klo);
}

__device__ __forceinline__ h16 load_b_frag_u8(const _Float16* p) {
  const int klo = ((threadIdx.x & 31) < 16) ? 0 : 16;
  h16 r;
#pragma unroll
  for (int i = 0; i < 4; ++i) {
    h4 v = *(const h4*)(p + klo + i * 4);
    r[i * 4 + 0] = v[0]; r[i * 4 + 1] = v[1];
    r[i * 4 + 2] = v[2]; r[i * 4 + 3] = v[3];
  }
  return r;
}

// ---------------------------------------------------------------------------
// Weight f32 -> f16 conversion (four 256x256 matrices)
// ---------------------------------------------------------------------------
__global__ __launch_bounds__(256) void cvt_weights_kernel(
    const float* __restrict__ a, const float* __restrict__ b,
    const float* __restrict__ c, const float* __restrict__ d,
    _Float16* __restrict__ ah, _Float16* __restrict__ bh,
    _Float16* __restrict__ ch, _Float16* __restrict__ dh) {
  int i = blockIdx.x * 256 + threadIdx.x;          // grid = 256 blocks -> 65536
  ah[i] = (_Float16)a[i];
  bh[i] = (_Float16)b[i];
  ch[i] = (_Float16)c[i];
  dh[i] = (_Float16)d[i];
}

// ---------------------------------------------------------------------------
// Channel LayerNorm: x [B,C,H,W] f32 -> out [B,N,C] f16 (token-major)
// One block = 32 pixels, full 256 channels. blockIdx.y selects x1/x2.
// ---------------------------------------------------------------------------
__global__ __launch_bounds__(256) void ln_kernel(
    const float* __restrict__ x1, const float* __restrict__ x2,
    const float* __restrict__ w1, const float* __restrict__ b1,
    const float* __restrict__ w2, const float* __restrict__ b2,
    _Float16* __restrict__ o1, _Float16* __restrict__ o2) {
  __shared__ float tile[C_DIM][33];   // [c][pixel], pad->conflict-free
  __shared__ float reds[8][32];
  __shared__ float redq[8][32];
  __shared__ float smean[32], srstd[32];

  const float* x;  const float* w;  const float* bb;  _Float16* o;
  if (blockIdx.y == 0) { x = x1; w = w1; bb = b1; o = o1; }
  else                 { x = x2; w = w2; bb = b2; o = o2; }

  const int pix0 = blockIdx.x * 32;          // 0..18431, never straddles batch
  const int b    = pix0 / NTOK;
  const int n0   = pix0 % NTOK;
  const int t    = threadIdx.x;
  const int p    = t & 31;
  const int cg   = t >> 5;

#pragma unroll
  for (int it = 0; it < 32; ++it) {
    int c = it * 8 + cg;
    tile[c][p] = x[((size_t)(b * C_DIM + c)) * NTOK + n0 + p];  // coalesced
  }
  __syncthreads();

  float s = 0.f, sq = 0.f;
#pragma unroll
  for (int i = 0; i < 32; ++i) {
    float v = tile[cg * 32 + i][p];
    s += v; sq += v * v;
  }
  reds[cg][p] = s;  redq[cg][p] = sq;
  __syncthreads();

  if (t < 32) {
    float ts = 0.f, tq = 0.f;
#pragma unroll
    for (int g = 0; g < 8; ++g) { ts += reds[g][t]; tq += redq[g][t]; }
    float mean = ts * (1.0f / 256.0f);
    float var  = tq * (1.0f / 256.0f) - mean * mean;
    smean[t] = mean;
    srstd[t] = rsqrtf(var + 1e-6f);
  }
  __syncthreads();

  const float wt = w[t], bt = bb[t];
  _Float16* obase = o + ((size_t)(b * NTOK + n0)) * C_DIM + t;
#pragma unroll 4
  for (int pp = 0; pp < 32; ++pp) {
    float v = (tile[t][pp] - smean[pp]) * srstd[pp] * wt + bt;
    obase[(size_t)pp * C_DIM] = (_Float16)v;    // 512B coalesced per pp
  }
}

// ---------------------------------------------------------------------------
// Projection GEMM: Out[r][o] = sum_c A[r][c] * W[o][c] + bias[o]   (f16 out)
// Block: 8 waves, tile M=64 x N=64; wave: 16x32 (2 D tiles), K-loop 8x32.
// ---------------------------------------------------------------------------
__device__ __forceinline__ void store_tile_f16(f8 acc, _Float16* tb /*[16][24]*/,
                                               _Float16* __restrict__ Out,
                                               int row0, int colbase) {
  const int lane = threadIdx.x & 31;
  const int lr   = lane & 15;
  const int aoff = (lane < 16) ? 0 : 8;
  // D layout: col o = lr, rows m = aoff + r.  Transpose through LDS.
#pragma unroll
  for (int r = 0; r < 8; ++r)
    tb[(aoff + r) * 24 + lr] = (_Float16)acc[r];
  // per-wave in-order LDS: store->load safe within the wave
  h8 v = *(const h8*)(tb + lr * 24 + ((lane >> 4) * 8));   // 48B rows, 16B ok
  *(h8*)(Out + (size_t)(row0 + lr) * C_DIM + colbase + ((lane >> 4) * 8)) = v;
}

__global__ __launch_bounds__(256) void gemm_proj_kernel(
    const _Float16* __restrict__ A,    // [B*N][256]
    const _Float16* __restrict__ Wh,   // [256][256] row-major [o][c]
    const float* __restrict__ bias,    // [256]
    _Float16* __restrict__ Out) {      // [B*N][256]
  __shared__ _Float16 tb[8][16][24];
  const int lane = threadIdx.x & 31;
  const int wid  = threadIdx.x >> 5;
  const int wm   = wid & 3, wn = wid >> 2;
  const int row0 = blockIdx.x * 64 + wm * 16;   // grid.x = 288
  const int nb   = blockIdx.y * 64 + wn * 32;   // grid.y = 4
  const int lr   = lane & 15;

  const _Float16* arow  = A  + (size_t)(row0 + lr) * C_DIM;
  const _Float16* brow0 = Wh + (size_t)(nb + lr) * C_DIM;
  const _Float16* brow1 = Wh + (size_t)(nb + 16 + lr) * C_DIM;

  f8 acc0, acc1;
  const float bv0 = bias[nb + lr], bv1 = bias[nb + 16 + lr];
#pragma unroll
  for (int i = 0; i < 8; ++i) { acc0[i] = bv0; acc1[i] = bv1; }

#pragma unroll
  for (int k0 = 0; k0 < C_DIM; k0 += 32) {
    h16 a   = load_a_frag(arow + k0);
    h16 bf0 = load_b_frag(brow0 + k0);
    h16 bf1 = load_b_frag(brow1 + k0);
    acc0 = wmma16(a, bf0, acc0);
    acc1 = wmma16(a, bf1, acc1);
  }
  store_tile_f16(acc0, &tb[wid][0][0], Out, row0, nb);
  store_tile_f16(acc1, &tb[wid][0][0], Out, row0, nb + 16);
}

// ---------------------------------------------------------------------------
// Flash attention. Block: 8 waves x 16 queries = 128 queries, grid (18, B).
// S^T = K*Q^T so per-query softmax stats are one scalar per lane.
// O^T accumulated as 16 f32 WMMA tiles (full C=256).
// ---------------------------------------------------------------------------
__global__ __launch_bounds__(256) void attention_kernel(
    const _Float16* __restrict__ Qh, const _Float16* __restrict__ Kh,
    const _Float16* __restrict__ Vh, _Float16* __restrict__ Oh) {
  __shared__ _Float16 Ks[32][264];      // K chunk rows, padded (conflict-free)
  __shared__ _Float16 VTs[256][40];     // V^T: [c][m], 80B rows (8B align)
  __shared__ _Float16 PTs[8][16][40];   // per-wave P^T in B-frag layout [n][m]
  __shared__ _Float16 tb[8][16][24];    // per-wave output transpose tile

  const int b    = blockIdx.y;
  const int q0   = blockIdx.x * 128;
  const int lane = threadIdx.x & 31;
  const int wid  = threadIdx.x >> 5;
  const int lr   = lane & 15;
  const int aoff = (lane < 16) ? 0 : 8;
  const int klo  = (lane < 16) ? 0 : 16;
  const int t    = threadIdx.x;
  const int qrow = q0 + wid * 16 + lr;

  // Q fragments in B layout (col n = lr = query), pre-scaled by ATT_SCALE.
  h16 qf[8];
  const _Float16* qbase = Qh + ((size_t)(b * NTOK + qrow)) * C_DIM;
#pragma unroll
  for (int kk = 0; kk < 8; ++kk) {
    h16 q = *(const h16*)(qbase + kk * 32 + klo);
#pragma unroll
    for (int i = 0; i < 16; ++i) q[i] = q[i] * (_Float16)ATT_SCALE;
    qf[kk] = q;
  }

  f8 oacc[16];
#pragma unroll
  for (int ct = 0; ct < 16; ++ct)
#pragma unroll
    for (int i = 0; i < 8; ++i) oacc[ct][i] = 0.f;

  float m_i = -1e30f, l_i = 0.f;

  for (int m0 = 0; m0 < NTOK; m0 += 32) {
    __syncthreads();
    // Cooperative stage: K chunk (row-major) + V chunk (transposed).
#pragma unroll
    for (int i = 0; i < 4; ++i) {
      int idx = i * 256 + t;                 // 1024 x 8 halves = 32x256
      int m   = idx >> 5;
      int c8  = (idx & 31) << 3;
      *(h8*)&Ks[m][c8] =
          *(const h8*)(Kh + ((size_t)(b * NTOK + m0 + m)) * C_DIM + c8);
      h8 v = *(const h8*)(Vh + ((size_t)(b * NTOK + m0 + m)) * C_DIM + c8);
#pragma unroll
      for (int j = 0; j < 8; ++j) VTs[c8 + j][m] = v[j];
    }
    __syncthreads();

    // S^T tiles (keys m0..+15 and m0+16..+31) over full C.
    f8 st0, st1;
#pragma unroll
    for (int i = 0; i < 8; ++i) { st0[i] = 0.f; st1[i] = 0.f; }
#pragma unroll
    for (int kk = 0; kk < 8; ++kk) {
      h16 a0 = load_a_frag(&Ks[lr][kk * 32]);
      h16 a1 = load_a_frag(&Ks[16 + lr][kk * 32]);
      st0 = wmma16(a0, qf[kk], st0);
      st1 = wmma16(a1, qf[kk], st1);
    }

    // Online softmax (per-lane query column; half-swap merges the two halves).
    float mx = st0[0];
#pragma unroll
    for (int i = 1; i < 8; ++i) mx = fmaxf(mx, st0[i]);
#pragma unroll
    for (int i = 0; i < 8; ++i) mx = fmaxf(mx, st1[i]);
    mx = fmaxf(mx, __shfl_xor(mx, 16, 32));
    float m_new = fmaxf(m_i, mx);
    float corr  = __builtin_exp2f((m_i - m_new) * LOG2E);
    float lsum  = 0.f;
    h8 p0v, p1v;
#pragma unroll
    for (int i = 0; i < 8; ++i) {
      float e0 = __builtin_exp2f((st0[i] - m_new) * LOG2E);
      float e1 = __builtin_exp2f((st1[i] - m_new) * LOG2E);
      lsum += e0 + e1;
      p0v[i] = (_Float16)e0;
      p1v[i] = (_Float16)e1;
    }
    lsum += __shfl_xor(lsum, 16, 32);
    l_i = l_i * corr + lsum;
    m_i = m_new;

#pragma unroll
    for (int ct = 0; ct < 16; ++ct)
#pragma unroll
      for (int i = 0; i < 8; ++i) oacc[ct][i] *= corr;

    // P^T -> LDS in B-fragment layout: row n = lr, m contiguous.
    {
      _Float16* pr = &PTs[wid][lr][0];
      *(h4*)(pr + aoff)          = h4{p0v[0], p0v[1], p0v[2], p0v[3]};
      *(h4*)(pr + aoff + 4)      = h4{p0v[4], p0v[5], p0v[6], p0v[7]};
      *(h4*)(pr + 16 + aoff)     = h4{p1v[0], p1v[1], p1v[2], p1v[3]};
      *(h4*)(pr + 16 + aoff + 4) = h4{p1v[4], p1v[5], p1v[6], p1v[7]};
    }
    h16 bp = load_b_frag_u8(&PTs[wid][lr][0]);   // in-order LDS within wave

    // O^T[c][n] += V^T (A) x P^T (B) for all 16 c-tiles.
#pragma unroll
    for (int ct = 0; ct < 16; ++ct) {
      h16 av = load_a_frag_u8(&VTs[ct * 16 + lr][0]);
      oacc[ct] = wmma16(av, bp, oacc[ct]);
    }
  }

  // Epilogue: divide by l_i, transpose per c-tile through LDS, packed stores.
  const float inv = 1.f / l_i;
  _Float16* tbw = &tb[wid][0][0];
  const size_t obase = ((size_t)(b * NTOK + q0 + wid * 16)) * C_DIM;
#pragma unroll
  for (int ct = 0; ct < 16; ++ct) {
    h8 ov;
#pragma unroll
    for (int r = 0; r < 8; ++r) ov[r] = (_Float16)(oacc[ct][r] * inv);
    *(h8*)(tbw + lr * 24 + aoff) = ov;           // [n=lr][c_local], 16B ok
    h8 rv = *(const h8*)(tbw + lr * 24 + ((lane >> 4) * 8));
    *(h8*)(Oh + obase + (size_t)lr * C_DIM + ct * 16 + ((lane >> 4) * 8)) = rv;
  }
}

// ---------------------------------------------------------------------------
// Final projection + residual: d_out[b][o][n] = sum_c Wp[o][c]*O[b][n][c]
//                                               + bp[o] + x1[b][o][n]
// D layout gives 8 consecutive tokens per lane -> contiguous f32 stores.
// ---------------------------------------------------------------------------
__global__ __launch_bounds__(256) void gemm_final_kernel(
    const _Float16* __restrict__ A,    // Oh [B*N][256]
    const _Float16* __restrict__ Wh,   // Wp f16 [o][c]
    const float* __restrict__ bias,    // bp
    const float* __restrict__ resid,   // x1 [B][C][N]
    float* __restrict__ Out) {         // [B][C][N]
  const int lane = threadIdx.x & 31;
  const int wid  = threadIdx.x >> 5;
  const int wm   = wid & 3, wn = wid >> 2;
  const int row0 = blockIdx.x * 64 + wm * 16;
  const int nb   = blockIdx.y * 64 + wn * 32;
  const int lr   = lane & 15;
  const int aoff = (lane < 16) ? 0 : 8;

  const _Float16* arow  = A  + (size_t)(row0 + lr) * C_DIM;
  const _Float16* brow0 = Wh + (size_t)(nb + lr) * C_DIM;
  const _Float16* brow1 = Wh + (size_t)(nb + 16 + lr) * C_DIM;

  f8 acc0, acc1;
  const float bv0 = bias[nb + lr], bv1 = bias[nb + 16 + lr];
#pragma unroll
  for (int i = 0; i < 8; ++i) { acc0[i] = bv0; acc1[i] = bv1; }

#pragma unroll
  for (int k0 = 0; k0 < C_DIM; k0 += 32) {
    h16 a   = load_a_frag(arow + k0);
    h16 bf0 = load_b_frag(brow0 + k0);
    h16 bf1 = load_b_frag(brow1 + k0);
    acc0 = wmma16(a, bf0, acc0);
    acc1 = wmma16(a, bf1, acc1);
  }

  const int bidx = row0 / NTOK;        // 64-row tiles never straddle batches
  const int nin  = row0 % NTOK;
#pragma unroll
  for (int tilei = 0; tilei < 2; ++tilei) {
    f8 acc = tilei ? acc1 : acc0;
    const int oc = nb + tilei * 16 + lr;
    const size_t off = ((size_t)(bidx * C_DIM + oc)) * NTOK + nin + aoff;
    f4 r0 = *(const f4*)(resid + off);
    f4 r1 = *(const f4*)(resid + off + 4);
    f4 v0, v1;
#pragma unroll
    for (int i = 0; i < 4; ++i) { v0[i] = acc[i] + r0[i]; v1[i] = acc[4 + i] + r1[i]; }
    *(f4*)(Out + off)     = v0;
    *(f4*)(Out + off + 4) = v1;
  }
}

// ---------------------------------------------------------------------------
extern "C" void kernel_launch(void* const* d_in, const int* in_sizes, int n_in,
                              void* d_out, int out_size, void* d_ws, size_t ws_size,
                              hipStream_t stream) {
  (void)in_sizes; (void)n_in; (void)out_size; (void)ws_size;

  const float* x1   = (const float*)d_in[0];
  const float* x2   = (const float*)d_in[1];
  const float* Wq   = (const float*)d_in[2];
  const float* bq   = (const float*)d_in[3];
  const float* Wk   = (const float*)d_in[4];
  const float* bk   = (const float*)d_in[5];
  const float* Wv   = (const float*)d_in[6];
  const float* bv   = (const float*)d_in[7];
  const float* Wp   = (const float*)d_in[8];
  const float* bp   = (const float*)d_in[9];
  const float* w_nq = (const float*)d_in[10];
  const float* b_nq = (const float*)d_in[11];
  const float* w_nk = (const float*)d_in[12];
  const float* b_nk = (const float*)d_in[13];
  float* out = (float*)d_out;

  const size_t ACT = (size_t)BATCH * NTOK * C_DIM * sizeof(_Float16); // 9.44MB
  const size_t WSZ = (size_t)C_DIM * C_DIM * sizeof(_Float16);       // 128KB
  char* ws = (char*)d_ws;
  _Float16* X1h = (_Float16*)(ws);            ws += ACT;
  _Float16* X2h = (_Float16*)(ws);            ws += ACT;
  _Float16* Qh  = (_Float16*)(ws);            ws += ACT;
  _Float16* Kh  = (_Float16*)(ws);            ws += ACT;
  _Float16* Vh  = (_Float16*)(ws);            ws += ACT;
  _Float16* Ohb = (_Float16*)(ws);            ws += ACT;
  _Float16* Wqh = (_Float16*)(ws);            ws += WSZ;
  _Float16* Wkh = (_Float16*)(ws);            ws += WSZ;
  _Float16* Wvh = (_Float16*)(ws);            ws += WSZ;
  _Float16* Wph = (_Float16*)(ws);            ws += WSZ;

  cvt_weights_kernel<<<256, 256, 0, stream>>>(Wq, Wk, Wv, Wp, Wqh, Wkh, Wvh, Wph);
  ln_kernel<<<dim3(576, 2), 256, 0, stream>>>(x1, x2, w_nq, b_nq, w_nk, b_nk,
                                              X1h, X2h);
  gemm_proj_kernel<<<dim3(288, 4), 256, 0, stream>>>(X1h, Wqh, bq, Qh);
  gemm_proj_kernel<<<dim3(288, 4), 256, 0, stream>>>(X2h, Wkh, bk, Kh);
  gemm_proj_kernel<<<dim3(288, 4), 256, 0, stream>>>(X2h, Wvh, bv, Vh);
  attention_kernel<<<dim3(18, BATCH), 256, 0, stream>>>(Qh, Kh, Vh, Ohb);
  gemm_final_kernel<<<dim3(288, 4), 256, 0, stream>>>(Ohb, Wph, bp, x1, out);
}
